// VGAE_37108517438028
// MI455X (gfx1250) — matrix operand order repaired and checked by reading, not compiled, not run on hardware
//
#include <hip/hip_runtime.h>
#include <cstddef>

// Problem constants (match reference)
#define NN 100000     // nodes
#define NE 1600000    // edges
#define EAtt 7        // edge attr dim
#define HD 64         // hidden
#define ZD 32         // latent
#define GG 512        // graphs
#define CC 6          // classes

typedef __attribute__((ext_vector_type(16))) _Float16 v16h;
typedef __attribute__((ext_vector_type(8)))  _Float16 v8h;
typedef __attribute__((ext_vector_type(8)))  float    v8f;

// ---------------------------------------------------------------------------
// Fragment-order LDS layout (contiguous k order, A and B alike), so each lane's WMMA
// operand is ONE contiguous 32-byte LDS read (2 x ds_load_b128):
//   A (16x32 f16) element (M, kl):  lane = M + 16*(kl>=16),
//                                   t    = kl&15
//   B (32x16 f16) element (kl, n):  lane = n + 16*(kl>=16), t = kl&15
//   C (16x16 f32): lane<16 -> M=r, N=lane;  lane>=16 -> M=r+8, N=lane-16
// A tiles: AshF[wave][ks][lane] (v16h);  B tiles: BshF[ks][nt][lane] (v16h)
// ---------------------------------------------------------------------------
__device__ inline void wave_gemm_frag(v16h (*Af)[32],     // [2][32] this wave
                                      v16h (*Bf)[4][32],  // [2][4][32]
                                      int lane, v8f acc[4]) {
#pragma unroll
  for (int ks = 0; ks < 2; ++ks) {
    v16h aF = Af[ks][lane];
#pragma unroll
    for (int nt = 0; nt < 4; ++nt) {
      v16h bF = Bf[ks][nt][lane];
      acc[nt] = __builtin_amdgcn_wmma_f32_16x16x32_f16(
          false, aF, false, bF, (short)0, acc[nt], false, false);
    }
  }
}

// Stage a row-major [64][64] f32 weight matrix into B fragment layout.
__device__ inline void stage_B64x64(const float* __restrict__ W,
                                    v16h (*Bf)[4][32], int tid) {
  for (int idx = tid; idx < HD * HD; idx += 256) {
    int k = idx >> 6, j = idx & 63;
    int ks = k >> 5, kl = k & 31;
    int nt = j >> 4, n = j & 15;
    int bl = n + ((kl >= 16) ? 16 : 0);
    ((_Float16*)&Bf[ks][nt][bl])[kl & 15] = (_Float16)W[idx];
  }
}

// Pack 32 consecutive-K values of one A row into fragment layout
// (4 contiguous 8-half groups -> 4 x 16B LDS stores).
__device__ inline void stage_A_row(v16h (*Af)[32], int ks, int M,
                                   const _Float16 vals[32]) {
#pragma unroll
  for (int g = 0; g < 4; ++g) {
    int lanep = M + 16 * (g >> 1);
    int t0 = (g & 1) ? 8 : 0;
    v8h pk;
#pragma unroll
    for (int i = 0; i < 8; ++i) pk[i] = vals[g * 8 + i];
    *(v8h*)((_Float16*)&Af[ks][lanep] + t0) = pk;
  }
}

// ---------------------------------------------------------------------------
__global__ void fill_zero_kernel(float4* __restrict__ p, long long n4) {
  long long i = (long long)blockIdx.x * blockDim.x + threadIdx.x;
  long long s = (long long)gridDim.x * blockDim.x;
  float4 z = {0.f, 0.f, 0.f, 0.f};
  for (; i < n4; i += s) p[i] = z;
}

// Layer-1 edge phase: enc = ea@W_e1 + b_e1 (scalar), m = relu(x[src]+enc),
// scatter-sum into agg1[dst].
__global__ void edge1_kernel(const float* __restrict__ x, const int* __restrict__ ei,
                             const float* __restrict__ ea, const float* __restrict__ W_e1,
                             const float* __restrict__ b_e1, float* __restrict__ agg1) {
  int e = blockIdx.x * blockDim.x + threadIdx.x;
  if (e >= NE) return;
  const float* a = ea + (size_t)e * EAtt;
  float enc = b_e1[0];
#pragma unroll
  for (int k = 0; k < EAtt; ++k) enc += a[k] * W_e1[k];
  int s = ei[e], d = ei[NE + e];
  float m = fmaxf(x[s] + enc, 0.f);
  atomicAdd(&agg1[d], m);
}

// Layer-1 node MLP: t = x+agg1; hid = relu(t*W1a+b1a); h1 = leaky(hid@W1b+b1b)
__global__ void __launch_bounds__(256) mlp1_kernel(
    const float* __restrict__ x, const float* __restrict__ agg1,
    const float* __restrict__ W1a, const float* __restrict__ b1a,
    const float* __restrict__ W1b, const float* __restrict__ b1b,
    float* __restrict__ h1) {
  __shared__ v16h BshF[2][4][32];
  __shared__ v16h AshF[8][2][32];
  const int tid = threadIdx.x, wave = tid >> 5, lane = tid & 31;
  stage_B64x64(W1b, BshF, tid);
  const int nodeBase = blockIdx.x * 128;
  {
    int nloc = tid >> 1;
    int node = nodeBase + nloc;
    int ks = tid & 1;
    int M = nloc & 15, wv = nloc >> 4;
    float t = (node < NN) ? (x[node] + agg1[node]) : 0.0f;
    _Float16 vals[32];
#pragma unroll
    for (int jj = 0; jj < 32; ++jj) {
      int j = ks * 32 + jj;
      vals[jj] = (_Float16)fmaxf(fmaf(t, W1a[j], b1a[j]), 0.f);
    }
    stage_A_row(AshF[wv], ks, M, vals);
  }
  __syncthreads();
  v8f acc[4] = {};
  wave_gemm_frag(AshF[wave], BshF, lane, acc);
  const int rowM = lane & 15, hi = (lane >= 16) ? 1 : 0;
  const int nodeTile = nodeBase + wave * 16;
  auto epi = [&](int guard) {
#pragma unroll
    for (int nt = 0; nt < 4; ++nt) {
      int j = nt * 16 + rowM;
      float bj = b1b[j];
#pragma unroll
      for (int r = 0; r < 8; ++r) {
        int node = nodeTile + r + hi * 8;
        if (!guard || node < NN) {
          float v = acc[nt][r] + bj;
          h1[(size_t)node * HD + j] = fmaxf(v, 0.1f * v);  // leaky_relu(v,0.1)
        }
      }
    }
  };
  if (nodeBase + 128 <= NN) epi(0); else epi(1);
}

// Layer-2 edge phase (the heavy one): per (edge, 8-channel group):
// enc_j = ea@W_e2[:,j]+b_e2[j]; m = relu(h1[src,j]+enc); atomic add agg2[dst,j]
__global__ void __launch_bounds__(256) edge2_kernel(
    const float* __restrict__ h1, const int* __restrict__ ei,
    const float* __restrict__ ea, const float* __restrict__ W_e2,
    const float* __restrict__ b_e2, float* __restrict__ agg2) {
  __shared__ float Wsh[EAtt * HD];
  __shared__ float bsh[HD];
  for (int i = threadIdx.x; i < EAtt * HD; i += blockDim.x) Wsh[i] = W_e2[i];
  if (threadIdx.x < HD) bsh[threadIdx.x] = b_e2[threadIdx.x];
  __syncthreads();
  long long gid = (long long)blockIdx.x * blockDim.x + threadIdx.x;
  if (gid >= (long long)NE * 8) return;
  int e = (int)(gid >> 3);
  int j0 = ((int)gid & 7) * 8;
  const float* a = ea + (size_t)e * EAtt;
  float a0 = a[0], a1 = a[1], a2 = a[2], a3 = a[3], a4 = a[4], a5 = a[5], a6 = a[6];
  int s = ei[e], d = ei[NE + e];
  const float4* hs = (const float4*)(h1 + (size_t)s * HD + j0);
  float4 h0 = hs[0], h1v = hs[1];
  float hv[8] = {h0.x, h0.y, h0.z, h0.w, h1v.x, h1v.y, h1v.z, h1v.w};
  float* ad = agg2 + (size_t)d * HD + j0;
#pragma unroll
  for (int jj = 0; jj < 8; ++jj) {
    int j = j0 + jj;
    float enc = bsh[j] + a0 * Wsh[0 * HD + j] + a1 * Wsh[1 * HD + j] +
                a2 * Wsh[2 * HD + j] + a3 * Wsh[3 * HD + j] +
                a4 * Wsh[4 * HD + j] + a5 * Wsh[5 * HD + j] + a6 * Wsh[6 * HD + j];
    float m = fmaxf(hv[jj] + enc, 0.f);
    atomicAdd(&ad[jj], m);
  }
}

// Layer-2 node MLP: hin = h1+agg2; hid = relu(hin@W2a+b2a);
// h2 = leaky(hid@W2b+b2b) written IN PLACE over agg2.
__global__ void __launch_bounds__(256) mlp2_kernel(
    const float* __restrict__ h1, float* __restrict__ agg2_h2,
    const float* __restrict__ W2a, const float* __restrict__ b2a,
    const float* __restrict__ W2b, const float* __restrict__ b2b) {
  __shared__ v16h BshA[2][4][32];
  __shared__ v16h BshB[2][4][32];
  __shared__ v16h AshF[8][2][32];
  const int tid = threadIdx.x, wave = tid >> 5, lane = tid & 31;
  stage_B64x64(W2a, BshA, tid);
  stage_B64x64(W2b, BshB, tid);
  const int nodeBase = blockIdx.x * 128;
  {
    int nloc = tid >> 1;
    int node = min(nodeBase + nloc, NN - 1);
    int ks = tid & 1;
    int M = nloc & 15, wv = nloc >> 4;
    const float* hrow = h1 + (size_t)node * HD;
    const float* arow = agg2_h2 + (size_t)node * HD;
    _Float16 vals[32];
#pragma unroll
    for (int jj = 0; jj < 32; ++jj) {
      int j = ks * 32 + jj;
      vals[jj] = (_Float16)(hrow[j] + arow[j]);
    }
    stage_A_row(AshF[wv], ks, M, vals);
  }
  __syncthreads();
  v8f acc[4] = {};
  wave_gemm_frag(AshF[wave], BshA, lane, acc);
  __syncthreads();
  const int rowM = lane & 15, hi = (lane >= 16) ? 1 : 0;
  // hid = relu(acc + b2a) written straight into A-fragment layout for GEMM2
#pragma unroll
  for (int nt = 0; nt < 4; ++nt) {
    int j = nt * 16 + rowM;
    float bj = b2a[j];
    int ks = j >> 5, kl = j & 31;
    int t = kl & 15;
    int laneOff = (kl & 16) ? 16 : 0;
#pragma unroll
    for (int r = 0; r < 8; ++r) {
      int M = r + hi * 8;
      float v = acc[nt][r] + bj;
      ((_Float16*)&AshF[wave][ks][M + laneOff])[t] = (_Float16)fmaxf(v, 0.f);
    }
  }
  __syncthreads();
  v8f acc2[4] = {};
  wave_gemm_frag(AshF[wave], BshB, lane, acc2);
  const int nodeTile = nodeBase + wave * 16;
  auto epi = [&](int guard) {
#pragma unroll
    for (int nt = 0; nt < 4; ++nt) {
      int j = nt * 16 + rowM;
      float bj = b2b[j];
#pragma unroll
      for (int r = 0; r < 8; ++r) {
        int node = nodeTile + r + hi * 8;
        if (!guard || node < NN) {
          float v = acc2[nt][r] + bj;
          agg2_h2[(size_t)node * HD + j] = fmaxf(v, 0.1f * v);
        }
      }
    }
  };
  if (nodeBase + 128 <= NN) epi(0); else epi(1);
}

// mu/logvar heads (fused as one 64x64 B: cols 0..31 = W_mu, 32..63 = W_lv),
// reparameterize, write z/mu/lv (nontemporal), atomic pool of z by graph.
__global__ void __launch_bounds__(256) latent_kernel(
    const float* __restrict__ h2, const float* __restrict__ eps,
    const int* __restrict__ batch,
    const float* __restrict__ W_mu, const float* __restrict__ b_mu,
    const float* __restrict__ W_lv, const float* __restrict__ b_lv,
    float* __restrict__ z_out, float* __restrict__ mu_out,
    float* __restrict__ lv_out, float* __restrict__ pool) {
  __shared__ v16h BshF[2][4][32];
  __shared__ v16h AshF[8][2][32];
  const int tid = threadIdx.x, wave = tid >> 5, lane = tid & 31;
  for (int idx = tid; idx < HD * ZD; idx += 256) {
    int k = idx >> 5, j = idx & 31;
    int ks = k >> 5, kl = k & 31;
    int nt = j >> 4, n = j & 15;
    int bl = n + ((kl >= 16) ? 16 : 0);
    int t = kl & 15;
    ((_Float16*)&BshF[ks][nt][bl])[t]     = (_Float16)W_mu[idx];  // cols 0..31
    ((_Float16*)&BshF[ks][2 + nt][bl])[t] = (_Float16)W_lv[idx];  // cols 32..63
  }
  const int nodeBase = blockIdx.x * 128;
  {
    int nloc = tid >> 1;
    int node = min(nodeBase + nloc, NN - 1);
    int ks = tid & 1;
    int M = nloc & 15, wv = nloc >> 4;
    const float* hrow = h2 + (size_t)node * HD;
    _Float16 vals[32];
#pragma unroll
    for (int jj = 0; jj < 32; ++jj) vals[jj] = (_Float16)hrow[ks * 32 + jj];
    stage_A_row(AshF[wv], ks, M, vals);
  }
  __syncthreads();
  v8f acc[4] = {};
  wave_gemm_frag(AshF[wave], BshF, lane, acc);
  const int rowM = lane & 15, hi = (lane >= 16) ? 1 : 0;
  const int nodeTile = nodeBase + wave * 16;
  auto epi = [&](int guard) {
#pragma unroll
    for (int t = 0; t < 2; ++t) {
      int j = t * 16 + rowM;  // latent channel 0..31
      float bm = b_mu[j], bl = b_lv[j];
#pragma unroll
      for (int r = 0; r < 8; ++r) {
        int node = nodeTile + r + hi * 8;
        if (!guard || node < NN) {
          float mu = acc[t][r] + bm;
          float lv = acc[2 + t][r] + bl;
          size_t o = (size_t)node * ZD + j;
          float zv = fmaf(eps[o], __expf(0.5f * lv), mu);
          __builtin_nontemporal_store(zv, &z_out[o]);
          __builtin_nontemporal_store(mu, &mu_out[o]);
          __builtin_nontemporal_store(lv, &lv_out[o]);
          atomicAdd(&pool[(size_t)batch[node] * ZD + j], zv);
        }
      }
    }
  };
  if (nodeBase + 128 <= NN) epi(0); else epi(1);
}

__global__ void count_kernel(const int* __restrict__ batch, float* __restrict__ cnt) {
  int i = blockIdx.x * blockDim.x + threadIdx.x;
  if (i < NN) atomicAdd(&cnt[batch[i]], 1.0f);
}

__global__ void classify_kernel(const float* __restrict__ pool, const float* __restrict__ cnt,
                                const float* __restrict__ W_cls, const float* __restrict__ b_cls,
                                float* __restrict__ logits) {
  int idx = blockIdx.x * blockDim.x + threadIdx.x;
  if (idx >= GG * CC) return;
  int g = idx / CC, c = idx % CC;
  float inv = 1.0f / fmaxf(cnt[g], 1.0f);
  float s = b_cls[c];
  const float* pg = pool + (size_t)g * ZD;
#pragma unroll
  for (int j = 0; j < ZD; ++j) s += (pg[j] * inv) * W_cls[j * CC + c];
  logits[idx] = s;
}

// ---------------------------------------------------------------------------
extern "C" void kernel_launch(void* const* d_in, const int* in_sizes, int n_in,
                              void* d_out, int out_size, void* d_ws, size_t ws_size,
                              hipStream_t stream) {
  (void)in_sizes; (void)n_in; (void)out_size; (void)ws_size;
  const float* x     = (const float*)d_in[0];
  const int*   ei    = (const int*)d_in[1];
  const float* ea    = (const float*)d_in[2];
  const int*   batch = (const int*)d_in[3];
  const float* eps   = (const float*)d_in[4];
  const float* W_e1  = (const float*)d_in[5];  const float* b_e1 = (const float*)d_in[6];
  const float* W1a   = (const float*)d_in[7];  const float* b1a  = (const float*)d_in[8];
  const float* W1b   = (const float*)d_in[9];  const float* b1b  = (const float*)d_in[10];
  const float* W_e2  = (const float*)d_in[11]; const float* b_e2 = (const float*)d_in[12];
  const float* W2a   = (const float*)d_in[13]; const float* b2a  = (const float*)d_in[14];
  const float* W2b   = (const float*)d_in[15]; const float* b2b  = (const float*)d_in[16];
  const float* W_mu  = (const float*)d_in[17]; const float* b_mu = (const float*)d_in[18];
  const float* W_lv  = (const float*)d_in[19]; const float* b_lv = (const float*)d_in[20];
  const float* W_cls = (const float*)d_in[21]; const float* b_cls= (const float*)d_in[22];

  float* out    = (float*)d_out;
  float* z_out  = out;
  float* mu_out = out + (size_t)NN * ZD;
  float* lv_out = out + 2 * (size_t)NN * ZD;
  float* logits = out + 3 * (size_t)NN * ZD;

  float* ws = (float*)d_ws;
  size_t o = 0;
  float* agg1 = ws + o; o += ((NN + 255) / 256) * 256;   // [N]
  float* h1   = ws + o; o += (size_t)NN * HD;            // [N,64]
  float* agg2 = ws + o; o += (size_t)NN * HD;            // [N,64] -> becomes h2
  float* pool = ws + o; o += (size_t)GG * ZD;            // [G,32]
  float* cnt  = ws + o; o += GG;                         // [G]
  const long long ws_elems = (long long)o;               // multiple of 4

  const int tb = 256;
  fill_zero_kernel<<<1024, tb, 0, stream>>>((float4*)ws, ws_elems / 4);
  edge1_kernel<<<(NE + tb - 1) / tb, tb, 0, stream>>>(x, ei, ea, W_e1, b_e1, agg1);
  mlp1_kernel<<<(NN + 127) / 128, tb, 0, stream>>>(x, agg1, W1a, b1a, W1b, b1b, h1);
  {
    long long work = (long long)NE * 8;
    edge2_kernel<<<(unsigned)((work + tb - 1) / tb), tb, 0, stream>>>(h1, ei, ea, W_e2, b_e2, agg2);
  }
  mlp2_kernel<<<(NN + 127) / 128, tb, 0, stream>>>(h1, agg2, W2a, b2a, W2b, b2b);
  latent_kernel<<<(NN + 127) / 128, tb, 0, stream>>>(agg2, eps, batch, W_mu, b_mu, W_lv, b_lv,
                                                     z_out, mu_out, lv_out, pool);
  count_kernel<<<(NN + tb - 1) / tb, tb, 0, stream>>>(batch, cnt);
  classify_kernel<<<(GG * CC + tb - 1) / tb, tb, 0, stream>>>(pool, cnt, W_cls, b_cls, logits);
}
